// DeepGCNLayer_38611755991861
// MI455X (gfx1250) — matrix operand order repaired and checked by reading, not compiled, not run on hardware
//
#include <hip/hip_runtime.h>
#include <hip/hip_bf16.h>

#define N_NODES 100000
#define N_EDGES 1600000
#define HDIM    128

typedef __attribute__((ext_vector_type(16))) _Float16 v16h;
typedef __attribute__((ext_vector_type(8)))  float    v8f;

// ---------------- degree / normalization ----------------

__global__ void deg_init_kernel(float* __restrict__ deg) {
    int i = blockIdx.x * blockDim.x + threadIdx.x;
    if (i < N_NODES) deg[i] = 1.0f;  // self loop contributes 1
}

__global__ void deg_edges_kernel(const int* __restrict__ ei, float* __restrict__ deg) {
    int e = blockIdx.x * blockDim.x + threadIdx.x;
    if (e < N_EDGES) {
        int d = ei[N_EDGES + e];  // dst row of edge_index
        atomicAdd(&deg[d], 1.0f);
    }
}

__global__ void dinv_kernel(float* __restrict__ deg) {
    int i = blockIdx.x * blockDim.x + threadIdx.x;
    if (i < N_NODES) deg[i] = rsqrtf(deg[i]);  // deg >= 1 always (self loop)
}

// ---------------- pre-pack W into WMMA B-fragment layout ----------------
// 32 fragments: (k0idx 0..3) x (nt 0..7). Per fragment, per lane: 8 dwords
// (16 f16) laid out exactly as V_WMMA_F32_16X16X32_F16 expects for SRC1:
//   lanes 0-15:  N = nt*16 + lane,    K = k0..k0+7 then k0+16..k0+23 (packed pairs)
//   lanes 16-31: N = nt*16 + lane-16, K = k0+8..k0+15 then k0+24..k0+31

__global__ void pack_w_kernel(const float* __restrict__ W, unsigned int* __restrict__ wp) {
    int t = blockIdx.x * blockDim.x + threadIdx.x;  // 0..1023
    if (t >= 1024) return;
    int frag = t >> 5;          // 0..31
    int lane = t & 31;
    int k0   = (frag >> 3) * 32;
    int nt   = frag & 7;
    int half = lane >> 4;
    int lrow = lane & 15;
    int col  = nt * 16 + lrow;
    int kbb  = k0 + half * 8;
    unsigned int* dst = wp + ((size_t)frag * 32 + lane) * 8;
    #pragma unroll
    for (int v = 0; v < 8; ++v) {
        _Float16 lo = (_Float16)W[(size_t)(kbb + (v >> 2) * 8 + 2 * v + 0) * HDIM + col];
        _Float16 hi = (_Float16)W[(size_t)(kbb + (v >> 2) * 8 + 2 * v + 1) * HDIM + col];
        unsigned short ulo, uhi;
        __builtin_memcpy(&ulo, &lo, 2);
        __builtin_memcpy(&uhi, &hi, 2);
        dst[v] = (unsigned int)ulo | ((unsigned int)uhi << 16);
    }
}

// ---------------- GEMM: h = x @ W via WMMA f16->f32 ----------------
// One wave computes a 16-row x 128-col strip of h.
// A (16x32 f16): lanes 0-15 -> M=lane, K=k0..k0+7 then k0+16..k0+23;
//                lanes 16-31 -> K=k0+8..k0+15 then k0+24..k0+31.
// B fragments come pre-packed from pack_w_kernel (2x b128 per lane).
// C/D (16x16 f32): VGPR j -> M = j + (lane>=16 ? 8 : 0), N = lane&15.

__global__ __launch_bounds__(256) void gemm_wmma_kernel(
    const float* __restrict__ x, const v16h* __restrict__ wfrag,
    float* __restrict__ h) {
    int wave = (blockIdx.x * blockDim.x + threadIdx.x) >> 5;
    int lane = threadIdx.x & 31;
    int m0 = wave * 16;
    if (m0 >= N_NODES) return;  // wave-uniform: EXEC stays all-ones inside

    int half = lane >> 4;   // 0 or 1
    int lrow = lane & 15;

    v8f acc[8];
    #pragma unroll
    for (int i = 0; i < 8; ++i) acc[i] = (v8f){0,0,0,0,0,0,0,0};

    const float4* xr = (const float4*)(x + (size_t)(m0 + lrow) * HDIM);

    #pragma unroll
    for (int k0i = 0; k0i < 4; ++k0i) {
        int k0 = k0i * 32;
        int kb = k0 + half * 8;           // multiple of 8
        // A tile: two contiguous 8-float runs -> 4x b128 loads
        float4 p0 = xr[kb / 4 + 0];
        float4 p1 = xr[kb / 4 + 1];
        float4 p2 = xr[kb / 4 + 4];
        float4 p3 = xr[kb / 4 + 5];
        v16h a;
        a[0]  = (_Float16)p0.x; a[1]  = (_Float16)p0.y;
        a[2]  = (_Float16)p0.z; a[3]  = (_Float16)p0.w;
        a[4]  = (_Float16)p1.x; a[5]  = (_Float16)p1.y;
        a[6]  = (_Float16)p1.z; a[7]  = (_Float16)p1.w;
        a[8]  = (_Float16)p2.x; a[9]  = (_Float16)p2.y;
        a[10] = (_Float16)p2.z; a[11] = (_Float16)p2.w;
        a[12] = (_Float16)p3.x; a[13] = (_Float16)p3.y;
        a[14] = (_Float16)p3.z; a[15] = (_Float16)p3.w;

        #pragma unroll
        for (int nt = 0; nt < 8; ++nt) {
            v16h b = wfrag[(size_t)(k0i * 8 + nt) * 32 + lane];  // 2x b128, L2-resident
            acc[nt] = __builtin_amdgcn_wmma_f32_16x16x32_f16(
                false, a, false, b, (short)0, acc[nt], false, false);
        }
    }

    // Store D tiles
    #pragma unroll
    for (int nt = 0; nt < 8; ++nt) {
        #pragma unroll
        for (int j = 0; j < 8; ++j) {
            int row = m0 + j + half * 8;
            int col = nt * 16 + lrow;
            h[(size_t)row * HDIM + col] = acc[nt][j];
        }
    }
}

// ---------------- self-loop init: out[i] = h[i] * dinv[i]^2 ----------------

__global__ void self_loop_kernel(const float* __restrict__ h,
                                 const float* __restrict__ dinv,
                                 float* __restrict__ out) {
    size_t idx = (size_t)blockIdx.x * blockDim.x + threadIdx.x;
    if (idx < (size_t)N_NODES * HDIM) {
        int node = (int)(idx >> 7);
        float di = dinv[node];
        out[idx] = h[idx] * di * di;
    }
}

// ---------------- edge scatter: out[dst] += h[src] * dinv[src]*dinv[dst] ----

__global__ __launch_bounds__(256) void scatter_kernel(
    const int* __restrict__ ei, const float* __restrict__ h,
    const float* __restrict__ dinv, float* __restrict__ out) {
    int wave = (blockIdx.x * blockDim.x + threadIdx.x) >> 5;
    int lane = threadIdx.x & 31;
    if (wave >= N_EDGES) return;
    int s = ei[wave];
    int d = ei[N_EDGES + wave];
    float norm = dinv[s] * dinv[d];
    const float4* hs = (const float4*)(h + (size_t)s * HDIM);
    float4 v = hs[lane];                    // coalesced 512B gather per wave
    float* od = out + (size_t)d * HDIM + lane * 4;
    atomicAdd(od + 0, v.x * norm);
    atomicAdd(od + 1, v.y * norm);
    atomicAdd(od + 2, v.z * norm);
    atomicAdd(od + 3, v.w * norm);
}

// ---------------- epilogue: +bias, exact GELU, LayerNorm ----------------

__global__ __launch_bounds__(256) void epilogue_kernel(
    float* __restrict__ out, const float* __restrict__ bias,
    const float* __restrict__ gamma, const float* __restrict__ beta) {
    int wave = (blockIdx.x * blockDim.x + threadIdx.x) >> 5;
    int lane = threadIdx.x & 31;
    if (wave >= N_NODES) return;
    float* row = out + (size_t)wave * HDIM;
    float v[4];
    #pragma unroll
    for (int i = 0; i < 4; ++i) {
        float t = row[lane * 4 + i] + bias[lane * 4 + i];
        // exact erf-form GELU
        v[i] = 0.5f * t * (1.0f + erff(t * 0.7071067811865476f));
    }
    // mean
    float s = v[0] + v[1] + v[2] + v[3];
    #pragma unroll
    for (int off = 16; off > 0; off >>= 1) s += __shfl_xor(s, off, 32);
    float mu = s * (1.0f / HDIM);
    // variance of (v - mu)
    float q = 0.0f;
    #pragma unroll
    for (int i = 0; i < 4; ++i) { float dlt = v[i] - mu; q += dlt * dlt; }
    #pragma unroll
    for (int off = 16; off > 0; off >>= 1) q += __shfl_xor(q, off, 32);
    float rstd = rsqrtf(q * (1.0f / HDIM) + 1e-5f);
    #pragma unroll
    for (int i = 0; i < 4; ++i) {
        int f = lane * 4 + i;
        row[f] = (v[i] - mu) * rstd * gamma[f] + beta[f];
    }
}

// ---------------- launch ----------------

extern "C" void kernel_launch(void* const* d_in, const int* in_sizes, int n_in,
                              void* d_out, int out_size, void* d_ws, size_t ws_size,
                              hipStream_t stream) {
    const float* x     = (const float*)d_in[0];
    const int*   ei    = (const int*)d_in[1];   // [2, E] int32
    const float* W     = (const float*)d_in[2];
    const float* bias  = (const float*)d_in[3];
    const float* gamma = (const float*)d_in[4];
    const float* beta  = (const float*)d_in[5];
    float* out = (float*)d_out;

    // workspace layout: dinv (N f32) | wpack (32 KB) | h (N*H f32)
    char* ws = (char*)d_ws;
    size_t off = 0;
    float* dinv = (float*)(ws + off);
    off += (((size_t)N_NODES * sizeof(float)) + 255) & ~(size_t)255;
    unsigned int* wpack = (unsigned int*)(ws + off);
    off += 32 * 32 * 8 * sizeof(unsigned int);  // 32 KB
    float* h = (float*)(ws + off);

    // 1) degrees + rsqrt normalization
    deg_init_kernel<<<(N_NODES + 255) / 256, 256, 0, stream>>>(dinv);
    deg_edges_kernel<<<(N_EDGES + 255) / 256, 256, 0, stream>>>(ei, dinv);
    dinv_kernel<<<(N_NODES + 255) / 256, 256, 0, stream>>>(dinv);

    // 2) pack W into WMMA B-fragment layout (one-shot, tiny)
    pack_w_kernel<<<4, 256, 0, stream>>>(W, wpack);

    // 3) h = x @ W  (6250 row-tiles of 16, one wave each)
    {
        int waves = (N_NODES + 15) / 16;
        int threads = waves * 32;
        gemm_wmma_kernel<<<(threads + 255) / 256, 256, 0, stream>>>(
            x, (const v16h*)wpack, h);
    }

    // 4) out = h * dinv^2 (self loops; also initializes d_out fully)
    {
        size_t total = (size_t)N_NODES * HDIM;
        self_loop_kernel<<<(unsigned)((total + 255) / 256), 256, 0, stream>>>(h, dinv, out);
    }

    // 5) edge scatter-add (one wave per edge)
    {
        size_t threads = (size_t)N_EDGES * 32;
        scatter_kernel<<<(unsigned)((threads + 255) / 256), 256, 0, stream>>>(ei, h, dinv, out);
    }

    // 6) bias + GELU + LayerNorm (one wave per node)
    {
        size_t threads = (size_t)N_NODES * 32;
        epilogue_kernel<<<(unsigned)((threads + 255) / 256), 256, 0, stream>>>(out, bias, gamma, beta);
    }
}
